// Conv4d1_8057358647868
// MI455X (gfx1250) — matrix operand order repaired and checked by reading, not compiled, not run on hardware
//
#include <hip/hip_runtime.h>

// ---------------------------------------------------------------------------
// 4D conv (3x3x3x3, pad 1) as im2col GEMM on CDNA5 WMMA.
//   M = Cout = 16, K = Cin*81 = 648 (padded 672 = 21 chunks of 32),
//   N = 16 output positions per tile; each wave owns <=2 tiles and keeps two
//   f32 accumulators so A-fragments/deltas are loaded once per K-chunk.
// fp32 accuracy via bf16 hi/lo split: x*w ~= xh*wh + xh*wl + xl*wh,
// accumulated in f32 by v_wmma_f32_16x16x32_bf16 (3 WMMAs per K-chunk/tile).
// Weight staging uses GLOBAL_LOAD_ASYNC_TO_LDS_B128 (ASYNCcnt) when available.
// ---------------------------------------------------------------------------

typedef __attribute__((ext_vector_type(16))) __bf16 v16bf;
typedef __attribute__((ext_vector_type(8)))  __bf16 v8bf;
typedef __attribute__((ext_vector_type(8)))  float  v8f;
typedef int v4i __attribute__((vector_size(16)));

typedef __attribute__((address_space(1))) v4i* gptr_v4i;   // global (__device__)
typedef __attribute__((address_space(3))) v4i* lptr_v4i;   // LDS (__shared__)

#define CIN   8
#define COUT  16
#define WDIM  32
#define HDIM  32
#define UDIM  20
#define VDIM  20
#define KTOT  648          // 8 * 81
#define KPAD  672          // 21 * 32
#define NCHUNK 21
#define HU    12           // u-halo extent for a 10-wide u half  (u in [-1,10])
#define HV    22           // v-halo extent                        (v in [-1,20])
#define PLANE (HU*HV)      // 264
#define XBLK  (CIN*3*3*PLANE)   // 19008 packed words = 76 KB
#define NTILE 13           // ceil(200/16); last tile half-masked

#if defined(__has_builtin)
#  if __has_builtin(__builtin_amdgcn_global_load_async_to_lds_b128)
#    define USE_ASYNC_LDS 1
#  endif
#endif
#ifndef USE_ASYNC_LDS
#  define USE_ASYNC_LDS 0
#endif

__device__ __forceinline__ void wait_async0() {
#if defined(__has_builtin) && __has_builtin(__builtin_amdgcn_s_wait_asynccnt)
    __builtin_amdgcn_s_wait_asynccnt(0);
#else
    asm volatile("s_wait_asynccnt 0x0" ::: "memory");
#endif
}

// Split fp32 into (bf16 hi, bf16 lo) with round-to-nearest-even, packed hi<<16|lo.
__device__ __forceinline__ unsigned pack_split(float v) {
    unsigned b  = __float_as_uint(v);
    unsigned hb = (b + 0x7FFFu + ((b >> 16) & 1u)) >> 16;
    float    hf = __uint_as_float(hb << 16);
    float    lf = v - hf;
    unsigned l32 = __float_as_uint(lf);
    unsigned lb  = (l32 + 0x7FFFu + ((l32 >> 16) & 1u)) >> 16;
    return (hb << 16) | (lb & 0xFFFFu);
}

__global__ __launch_bounds__(256, 1)
void conv4d_wmma_kernel(const float* __restrict__ x,
                        const float* __restrict__ wgt,
                        const float* __restrict__ bias,
                        float* __restrict__ out) {
    // ---- LDS: ~122 KB total -------------------------------------------------
    __shared__ __align__(16) unsigned xpk[XBLK];        // staged weights, then packed input
    __shared__ __align__(16) __bf16   Whi[COUT * KPAD]; // weight hi plane (SoA, K-major)
    __shared__ __align__(16) __bf16   Wlo[COUT * KPAD]; // weight lo plane
    __shared__ __align__(16) int      dlt[KPAD];        // im2col delta (word offsets)
    __shared__ float                  biasS[COUT];

    const int tid = threadIdx.x;            // 0..255, 8 wave32
    const int bid = blockIdx.x;             // 2*32*32*2 = 4096 blocks
    const int b0  = bid >> 11;
    const int w0  = (bid >> 6) & 31;
    const int h0  = (bid >> 1) & 31;
    const int uh  = bid & 1;                // which u-half: u in [10*uh, 10*uh+10)

    if (tid < COUT) biasS[tid] = bias[tid];

    // ---- stage raw weights via async global->LDS copy (contiguous [co][K]) --
#if USE_ASYNC_LDS
    for (int j = tid; j < (COUT * KTOT) / 4; j += 256) {
        __builtin_amdgcn_global_load_async_to_lds_b128(
            (gptr_v4i)(wgt + 4 * j),
            (lptr_v4i)(&xpk[4 * j]),
            0, 0);
    }
    wait_async0();
    __syncthreads();
#endif

    // ---- split weights into bf16 hi/lo planes -------------------------------
    for (int i = tid; i < COUT * KPAD; i += 256) {
        const int K  = i % KPAD;
        const int co = i / KPAD;
        float wv = 0.0f;
        if (K < KTOT) {
#if USE_ASYNC_LDS
            wv = __uint_as_float(xpk[co * KTOT + K]);
#else
            wv = wgt[co * KTOT + K];
#endif
        }
        unsigned p = pack_split(wv);
        Whi[i] = __builtin_bit_cast(__bf16, (unsigned short)(p >> 16));
        Wlo[i] = __builtin_bit_cast(__bf16, (unsigned short)(p & 0xFFFFu));
    }

    // ---- im2col delta table: K -> word offset inside xpk --------------------
    for (int K = tid; K < KPAD; K += 256) {
        int d = 0;                           // padded K: clamp to a valid addr
        if (K < KTOT) {                      // (weights are zero there anyway)
            int ci = K / 81, r = K % 81;
            int k1 = r / 27, k2 = (r / 9) % 3, k3 = (r / 3) % 3, k4 = r % 3;
            d = ((ci * 3 + k1) * 3 + k2) * PLANE + k3 * HV + k4;
        }
        dlt[K] = d;
    }

#if USE_ASYNC_LDS
    __syncthreads();   // weight reads of xpk done before x overwrites it
#endif

    // ---- stage haloed input block, split+packed -----------------------------
    for (int i = tid; i < XBLK; i += 256) {
        int hv = i % HV;  int t = i / HV;
        int hu = t % HU;  t /= HU;
        int dh = t % 3;   t /= 3;
        int dw = t % 3;   int ci = t / 3;
        int gw = w0 + dw - 1;
        int gh = h0 + dh - 1;
        int gu = uh * 10 + hu - 1;
        int gv = hv - 1;
        float v = 0.0f;
        if ((unsigned)gw < (unsigned)WDIM && (unsigned)gh < (unsigned)HDIM &&
            (unsigned)gu < (unsigned)UDIM && (unsigned)gv < (unsigned)VDIM)
            v = x[((((b0 * CIN + ci) * WDIM + gw) * HDIM + gh) * UDIM + gu) * VDIM + gv];
        xpk[i] = pack_split(v);
    }
    __syncthreads();

    // ---- per-wave WMMA: 2 tiles per wave, kc-outer --------------------------
    const int lane = tid & 31;
    const int wave = tid >> 5;
    const int mn   = lane & 15;     // A row (co) / B col (position) / D col
    const int half = lane >> 4;     // which 16-lane half

    const int  t1    = wave + 8;            // tile 0 is `wave` (always live)
    const bool has1  = (t1 < NTILE);        // wave-uniform

    const int pos0  = wave * 16 + mn;       // < 128+16 -> always valid
    const int lu0   = pos0 / VDIM;
    const int vn0   = pos0 % VDIM;
    const int base0 = lu0 * HV + vn0;

    const int  pos1  = t1 * 16 + mn;
    const bool live1 = has1 && (pos1 < 200);
    const int  cpos1 = (pos1 < 200) ? pos1 : 199;   // clamp for address gen
    const int  lu1   = cpos1 / VDIM;
    const int  vn1   = cpos1 % VDIM;
    const int  base1 = lu1 * HV + vn1;

    v8f acc0 = {0.f, 0.f, 0.f, 0.f, 0.f, 0.f, 0.f, 0.f};
    v8f acc1 = {0.f, 0.f, 0.f, 0.f, 0.f, 0.f, 0.f, 0.f};

    for (int kc = 0; kc < NCHUNK; ++kc) {
        // A fragment: contiguous 16-elem K run per lane (matches B:
        // lanes0-15 K=[0..16), lanes16-31 K=[16..32))
        const int ka = kc * 32 + half * 16;
        union { v16bf v; v8bf h[2]; } ahi, alo;
        ahi.h[0] = *(const v8bf*)&Whi[mn * KPAD + ka];
        ahi.h[1] = *(const v8bf*)&Whi[mn * KPAD + ka + 8];
        alo.h[0] = *(const v8bf*)&Wlo[mn * KPAD + ka];
        alo.h[1] = *(const v8bf*)&Wlo[mn * KPAD + ka + 8];

        // B-fragment deltas: contiguous 16-elem K run per lane (same order as A:
        // lanes0-15 K=[0..16), lanes16-31 K=[16..32)) — tile-invariant
        const int kb = kc * 32 + half * 16;
        const int4* dp = (const int4*)&dlt[kb];      // 64B-aligned
        int4 q0 = dp[0], q1 = dp[1], q2 = dp[2], q3 = dp[3];
        int dd[16] = {q0.x, q0.y, q0.z, q0.w, q1.x, q1.y, q1.z, q1.w,
                      q2.x, q2.y, q2.z, q2.w, q3.x, q3.y, q3.z, q3.w};

        // ---- tile 0 ----
        {
            unsigned p[16];
            #pragma unroll
            for (int e = 0; e < 16; ++e) p[e] = xpk[base0 + dd[e]];
            union { v16bf v; unsigned w[8]; } bhi, blo;
            #pragma unroll
            for (int j = 0; j < 8; ++j) {       // v_perm_b32: split hi/lo pairs
                bhi.w[j] = __builtin_amdgcn_perm(p[2*j+1], p[2*j], 0x07060302u);
                blo.w[j] = __builtin_amdgcn_perm(p[2*j+1], p[2*j], 0x05040100u);
            }
            acc0 = __builtin_amdgcn_wmma_f32_16x16x32_bf16(
                       false, ahi.v, false, bhi.v, (short)0, acc0, false, false);
            acc0 = __builtin_amdgcn_wmma_f32_16x16x32_bf16(
                       false, ahi.v, false, blo.v, (short)0, acc0, false, false);
            acc0 = __builtin_amdgcn_wmma_f32_16x16x32_bf16(
                       false, alo.v, false, bhi.v, (short)0, acc0, false, false);
        }

        // ---- tile 1 (wave-uniform branch: EXEC all-ones inside) ----
        if (has1) {
            unsigned p[16];
            #pragma unroll
            for (int e = 0; e < 16; ++e) p[e] = xpk[base1 + dd[e]];
            union { v16bf v; unsigned w[8]; } bhi, blo;
            #pragma unroll
            for (int j = 0; j < 8; ++j) {
                bhi.w[j] = __builtin_amdgcn_perm(p[2*j+1], p[2*j], 0x07060302u);
                blo.w[j] = __builtin_amdgcn_perm(p[2*j+1], p[2*j], 0x05040100u);
            }
            acc1 = __builtin_amdgcn_wmma_f32_16x16x32_bf16(
                       false, ahi.v, false, bhi.v, (short)0, acc1, false, false);
            acc1 = __builtin_amdgcn_wmma_f32_16x16x32_bf16(
                       false, ahi.v, false, blo.v, (short)0, acc1, false, false);
            acc1 = __builtin_amdgcn_wmma_f32_16x16x32_bf16(
                       false, alo.v, false, bhi.v, (short)0, acc1, false, false);
        }
    }

    // ---- epilogue: D layout lane L, VGPR r -> (co = r + 8*half, n = L&15) ---
    {
        const int gu0 = uh * 10 + lu0;
        #pragma unroll
        for (int r = 0; r < 8; ++r) {
            const int co = r + half * 8;
            out[((((b0 * COUT + co) * WDIM + w0) * HDIM + h0) * UDIM + gu0) * VDIM + vn0]
                = acc0[r] + biasS[co];
        }
    }
    if (live1) {
        const int gu1 = uh * 10 + lu1;
        #pragma unroll
        for (int r = 0; r < 8; ++r) {
            const int co = r + half * 8;
            out[((((b0 * COUT + co) * WDIM + w0) * HDIM + h0) * UDIM + gu1) * VDIM + vn1]
                = acc1[r] + biasS[co];
        }
    }
}

extern "C" void kernel_launch(void* const* d_in, const int* in_sizes, int n_in,
                              void* d_out, int out_size, void* d_ws, size_t ws_size,
                              hipStream_t stream) {
    (void)in_sizes; (void)n_in; (void)out_size; (void)d_ws; (void)ws_size;
    const float* x = (const float*)d_in[0];
    const float* w = (const float*)d_in[1];
    const float* b = (const float*)d_in[2];
    float* out     = (float*)d_out;
    dim3 grid(2 * 32 * 32 * 2);   // (b, w, h, u-half)
    dim3 block(256);              // 8 wave32
    hipLaunchKernelGGL(conv4d_wmma_kernel, grid, block, 0, stream, x, w, b, out);
}
